// ChangeDecoder_38560216383920
// MI455X (gfx1250) — matrix operand order repaired and checked by reading, not compiled, not run on hardware
//
#include <hip/hip_runtime.h>
#include <stdint.h>

// ---------------------------------------------------------------------------
// ChangeDecoder for MI455X (gfx1250, wave32). All channel contractions run on
// V_WMMA_F32_16X16X32_F16 with tiles staged cache->LDS via
// GLOBAL_LOAD_ASYNC_TO_LDS_B128 (ASYNCcnt); fp32 elementwise / LN / scan.
// ---------------------------------------------------------------------------

typedef __attribute__((ext_vector_type(16))) _Float16 v16h;
typedef __attribute__((ext_vector_type(8)))  float    v8f;

#define BN 4            // batch
#define HID 128
#define DIN 256         // D_INNER
#define DST 16          // D_STATE
#define DTR 8           // DT_RANK

// ---------------------------------------------------------------------------
// Activations
// ---------------------------------------------------------------------------
__device__ __forceinline__ float act_apply(float v, int act) {
  switch (act) {
    case 1: return v / (1.f + __expf(-v));                                   // silu
    case 2: { float c = 0.7978845608028654f * (v + 0.044715f * v * v * v);
              return 0.5f * v * (1.f + tanhf(c)); }                          // gelu(tanh)
    case 3: return v > 0.f ? v : 0.f;                                        // relu
    case 4: return v > 20.f ? v : log1pf(__expf(v));                         // softplus
    default: return v;
  }
}

// Issue one 16-byte async global->LDS copy (per lane). ASYNCcnt-tracked.
__device__ __forceinline__ void async_b128(unsigned lds_addr, uint64_t gaddr) {
  asm volatile("global_load_async_to_lds_b128 %0, %1, off"
               :: "v"(lds_addr), "v"(gaddr) : "memory");
}
__device__ __forceinline__ void wait_async0() {
  asm volatile("s_wait_asynccnt 0x0" ::: "memory");
}

// ---------------------------------------------------------------------------
// Batched WMMA GEMM:  C[z] = act( (A[z] (MxK) . W[z]^T (NxK) + bias[z]) * g + beta )
// A row-major lda, W row-major (N rows of K), C row-major ldc.
// Block: 128 threads = 4 waves; tile 64x64; each wave owns a 32x32 quadrant
// (2 A-frags x 2 B-frags -> 4 independent v_wmma per 32-wide K step).
// Tiles are staged as raw f32 in LDS (async path when interior), converted to
// f16 at fragment read.
// ---------------------------------------------------------------------------
__global__ __launch_bounds__(128)
void k_gemm_wmma(const float* __restrict__ A, long long lda, long long strideA,
                 const float* __restrict__ Wt, long long ldw, long long strideW,
                 float* __restrict__ C, long long ldc, long long strideC,
                 int M, int N, int Kd,
                 const float* __restrict__ bias, long long strideBias,
                 const float* __restrict__ g, const float* __restrict__ beta,
                 int act)
{
  // 36-float (144B) row stride: 16B-aligned chunks; 36*r mod 64 distinct for
  // r=0..15 -> conflict-free fragment reads.
  __shared__ __align__(16) float As32[64][36];
  __shared__ __align__(16) float Bs32[64][36];

  const int z    = blockIdx.z;
  const int m0   = blockIdx.y * 64;
  const int n0   = blockIdx.x * 64;
  const int tid  = threadIdx.x;
  const int lane = tid & 31;
  const int wave = tid >> 5;
  const int wy   = (wave >> 1) * 32;   // wave quadrant row (0/32)
  const int wx   = (wave & 1) * 32;    // wave quadrant col (0/32)
  const int l16  = lane & 15;
  const int hi   = lane >> 4;          // lane half of the wave

  const float* Az = A  + (long long)z * strideA;
  const float* Wz = Wt + (long long)z * strideW;

  v8f acc[2][2];
#pragma unroll
  for (int i = 0; i < 2; ++i)
#pragma unroll
    for (int j = 0; j < 2; ++j) acc[i][j] = v8f{0.f,0.f,0.f,0.f,0.f,0.f,0.f,0.f};

  // Cooperative loader: each thread fills one LDS row (lr) over 16 K-columns.
  const int lr = tid >> 1;          // 0..63
  const int lc = (tid & 1) * 16;    // 0 or 16

  const bool fastA = (m0 + 64 <= M) && ((lda & 3) == 0);
  const bool fastB = (n0 + 64 <= N) && ((ldw & 3) == 0);

  const unsigned ldsA = (unsigned)(uintptr_t)&As32[lr][lc];
  const unsigned ldsB = (unsigned)(uintptr_t)&Bs32[lr][lc];

  // LDS(f32) -> f16 fragment packers per CDNA5 16-bit WMMA VGPR layouts.
  auto ldA = [&](int r0) {
    v16h a;
    const int m  = r0 + l16;
    const int kb = hi * 8;
#pragma unroll
    for (int e = 0; e < 8; ++e) {
      a[e]     = (_Float16)As32[m][kb + e];        // K = kb+e      (VGPR0-3)
      a[8 + e] = (_Float16)As32[m][16 + kb + e];   // K = 16+kb+e   (VGPR4-7)
    }
    return a;
  };
  auto ldB = [&](int c0) {
    v16h b;
    const int n   = c0 + l16;
    const int kb2 = hi * 8;
#pragma unroll
    for (int e = 0; e < 8; ++e) {
      b[e]     = (_Float16)Bs32[n][kb2 + e];
      b[8 + e] = (_Float16)Bs32[n][16 + kb2 + e];
    }
    return b;
  };

  for (int kk = 0; kk < Kd; kk += 32) {
    const bool kfull = (kk + 32 <= Kd);
    bool used_async = false;
    // ---- A tile (64 x 32 f32) ----
    if (fastA && kfull) {
      uint64_t ga = (uint64_t)(Az + (long long)(m0 + lr) * lda + kk + lc);
      async_b128(ldsA,       ga);
      async_b128(ldsA + 16u, ga + 16u);
      async_b128(ldsA + 32u, ga + 32u);
      async_b128(ldsA + 48u, ga + 48u);
      used_async = true;
    } else {
      const int m  = m0 + lr;
      const int mc = m < (M - 1) ? m : (M - 1);
      float tmp[16];
#pragma unroll
      for (int j = 0; j < 16; ++j) {
        int k  = kk + lc + j;
        int kc = k < (Kd - 1) ? k : (Kd - 1);
        tmp[j] = Az[(long long)mc * lda + kc];       // always-valid address
      }
#pragma unroll
      for (int j = 0; j < 16; ++j) {
        int k = kk + lc + j;
        As32[lr][lc + j] = (m < M && k < Kd) ? tmp[j] : 0.f;   // select, no branch
      }
    }
    // ---- B tile (64 x 32 f32) ----
    if (fastB && kfull) {
      uint64_t gb = (uint64_t)(Wz + (long long)(n0 + lr) * ldw + kk + lc);
      async_b128(ldsB,       gb);
      async_b128(ldsB + 16u, gb + 16u);
      async_b128(ldsB + 32u, gb + 32u);
      async_b128(ldsB + 48u, gb + 48u);
      used_async = true;
    } else {
      const int n  = n0 + lr;
      const int nc = n < (N - 1) ? n : (N - 1);
      float tmp[16];
#pragma unroll
      for (int j = 0; j < 16; ++j) {
        int k  = kk + lc + j;
        int kc = k < (Kd - 1) ? k : (Kd - 1);
        tmp[j] = Wz[(long long)nc * ldw + kc];
      }
#pragma unroll
      for (int j = 0; j < 16; ++j) {
        int k = kk + lc + j;
        Bs32[lr][lc + j] = (n < N && k < Kd) ? tmp[j] : 0.f;
      }
    }
    if (used_async) wait_async0();   // drain this wave's async copies
    __syncthreads();

    v16h a0 = ldA(wy), a1 = ldA(wy + 16);
    v16h b0 = ldB(wx), b1 = ldB(wx + 16);
    acc[0][0] = __builtin_amdgcn_wmma_f32_16x16x32_f16(false, a0, false, b0, (short)0, acc[0][0], false, false);
    acc[0][1] = __builtin_amdgcn_wmma_f32_16x16x32_f16(false, a0, false, b1, (short)0, acc[0][1], false, false);
    acc[1][0] = __builtin_amdgcn_wmma_f32_16x16x32_f16(false, a1, false, b0, (short)0, acc[1][0], false, false);
    acc[1][1] = __builtin_amdgcn_wmma_f32_16x16x32_f16(false, a1, false, b1, (short)0, acc[1][1], false, false);
    __syncthreads();
  }

  // Epilogue. D layout: VGPR r -> M = r (lanes 0-15) / 8+r (lanes 16-31); N = lane%16.
  float* Cz = C + (long long)z * strideC;
#pragma unroll
  for (int rx = 0; rx < 2; ++rx) {
    const int n = n0 + wx + rx * 16 + l16;
    if (n >= N) continue;
    float bv = bias ? bias[(long long)z * strideBias + n] : 0.f;
    float gv = 1.f, bt = 0.f;
    if (g) { gv = g[n]; bt = beta[n]; }
#pragma unroll
    for (int ry = 0; ry < 2; ++ry) {
#pragma unroll
      for (int r = 0; r < 8; ++r) {
        int m = m0 + wy + ry * 16 + hi * 8 + r;
        if (m < M) {
          float v = acc[ry][rx][r] + bv;
          v = v * gv + bt;
          Cz[(long long)m * ldc + n] = act_apply(v, act);
        }
      }
    }
  }
}

// ---------------------------------------------------------------------------
// Elementwise / data-movement kernels
// ---------------------------------------------------------------------------
__global__ void k_fill(float* p, float v, long long n) {
  long long i = (long long)blockIdx.x * blockDim.x + threadIdx.x;
  if (i < n) p[i] = v;
}

__global__ void k_add(float* out, const float* a, const float* b, long long n) {
  long long i = (long long)blockIdx.x * blockDim.x + threadIdx.x;
  if (i < n) out[i] = a[i] + b[i];
}

__global__ void k_add_relu(float* out, const float* a, const float* b, long long n) {
  long long i = (long long)blockIdx.x * blockDim.x + threadIdx.x;
  if (i < n) { float v = a[i] + b[i]; out[i] = v > 0.f ? v : 0.f; }
}

__global__ void k_cat_ch(const float* pre, const float* post, float* out,
                         int Cc, int S, long long n) {   // out (B,2C,S)
  long long i = (long long)blockIdx.x * blockDim.x + threadIdx.x;
  if (i >= n) return;
  int s = (int)(i % S);
  int c = (int)((i / S) % (2 * Cc));
  long long b = i / ((long long)S * 2 * Cc);
  out[i] = (c < Cc) ? pre [(b * Cc + c)        * S + s]
                    : post[(b * Cc + (c - Cc)) * S + s];
}

__global__ void k_interleave_w(const float* pre, const float* post, float* out,
                               int Cc, int H, int W, long long n) { // out (B,C,H,2W)
  long long i = (long long)blockIdx.x * blockDim.x + threadIdx.x;
  if (i >= n) return;
  int W2 = 2 * W;
  int w2 = (int)(i % W2);
  int h  = (int)((i / W2) % H);
  int c  = (int)((i / ((long long)W2 * H)) % Cc);
  long long b = i / ((long long)W2 * H * Cc);
  const float* src = (w2 & 1) ? post : pre;
  out[i] = src[((b * Cc + c) * H + h) * (long long)W + (w2 >> 1)];
}

__global__ void k_cat_w(const float* pre, const float* post, float* out,
                        int Cc, int H, int W, long long n) {        // out (B,C,H,2W)
  long long i = (long long)blockIdx.x * blockDim.x + threadIdx.x;
  if (i >= n) return;
  int W2 = 2 * W;
  int w2 = (int)(i % W2);
  int h  = (int)((i / W2) % H);
  int c  = (int)((i / ((long long)W2 * H)) % Cc);
  long long b = i / ((long long)W2 * H * Cc);
  out[i] = (w2 < W) ? pre [((b * Cc + c) * H + h) * (long long)W + w2]
                    : post[((b * Cc + c) * H + h) * (long long)W + (w2 - W)];
}

__global__ void k_dwconv3(const float* __restrict__ in, const float* __restrict__ wgt,
                          const float* __restrict__ bias, float* __restrict__ out,
                          int Cc, int H, int W, long long total, int act) {
  long long i = (long long)blockIdx.x * blockDim.x + threadIdx.x;
  if (i >= total) return;
  int w = (int)(i % W);
  int h = (int)((i / W) % H);
  int c = (int)((i / ((long long)W * H)) % Cc);
  long long b = i / ((long long)W * H * Cc);
  const float* p = in + ((b * Cc + c) * (long long)H) * W;
  float s = bias ? bias[c] : 0.f;
#pragma unroll
  for (int ky = 0; ky < 3; ++ky) {
    int y = h + ky - 1;
    if (y < 0 || y >= H) continue;
#pragma unroll
    for (int kx = 0; kx < 3; ++kx) {
      int x = w + kx - 1;
      if (x < 0 || x >= W) continue;
      s += p[(long long)y * W + x] * wgt[c * 9 + ky * 3 + kx];
    }
  }
  out[i] = act_apply(s, act);
}

__global__ void k_nchw2nhwc(const float* in, float* out, int Cc, int H, int W, long long n) {
  long long i = (long long)blockIdx.x * blockDim.x + threadIdx.x;
  if (i >= n) return;
  int c = (int)(i % Cc);
  long long m = i / Cc;
  int w = (int)(m % W);
  int h = (int)((m / W) % H);
  long long b = m / ((long long)W * H);
  out[i] = in[((b * Cc + c) * (long long)H + h) * W + w];
}

__global__ void k_nhwc2nchw(const float* in, float* out, int Cc, int H, int W, long long n) {
  long long i = (long long)blockIdx.x * blockDim.x + threadIdx.x;
  if (i >= n) return;
  int w = (int)(i % W);
  int h = (int)((i / W) % H);
  int c = (int)((i / ((long long)W * H)) % Cc);
  long long b = i / ((long long)W * H * Cc);
  out[i] = in[((b * (long long)H + h) * W + w) * Cc + c];
}

// x-half of xz rows (R x 512) -> NCHW (B,256,H,W)
__global__ void k_split_x_nchw(const float* xz, float* xc, int H, int W, long long n) {
  long long i = (long long)blockIdx.x * blockDim.x + threadIdx.x;
  if (i >= n) return;
  int w = (int)(i % W);
  int h = (int)((i / W) % H);
  int d = (int)((i / ((long long)W * H)) % DIN);
  long long b = i / ((long long)W * H * DIN);
  xc[i] = xz[((b * H + h) * (long long)W + w) * (2 * DIN) + d];
}

// conv NCHW (B,256,H,W) -> xs (B,2,L,256): dir0 raster order, dir1 transposed order
__global__ void k_build_xs(const float* conv, float* xs, int H, int W, long long n) {
  long long i = (long long)blockIdx.x * blockDim.x + threadIdx.x;
  if (i >= n) return;
  long long L = (long long)H * W;
  int d = (int)(i % DIN);
  long long l = (i / DIN) % L;
  long long b = i / (DIN * L);
  int h = (int)(l / W), w = (int)(l % W);
  float v = conv[((b * DIN + d) * (long long)H + h) * W + w];
  xs[((b * 2 + 0) * L + l) * DIN + d] = v;
  long long m = (long long)w * H + h;
  xs[((b * 2 + 1) * L + m) * DIN + d] = v;
}

// Selective scan for one direction k. 16 lanes cooperate over the state dim (n),
// sequence walked forward or backward; output accumulated into y in raster order.
__global__ void k_scan(const float* __restrict__ delta,
                       const float* __restrict__ u, long long u_bstride,
                       const float* __restrict__ xdbl,
                       const float* __restrict__ A_logs,
                       const float* __restrict__ Ds,
                       float* __restrict__ y,
                       int L, int H, int W, int rev, int trans)
{
  int t = blockIdx.x * blockDim.x + threadIdx.x;
  int n = t & 15;
  int d = (t >> 4) & (DIN - 1);
  int b = t >> 12;
  if (b >= BN) return;
  float An = -__expf(A_logs[d * DST + n]);
  float Dd = Ds[d];
  const float* dl = delta + (long long)b * L * DIN + d;
  const float* ul = u     + (long long)b * u_bstride + d;
  const float* xb = xdbl  + (long long)b * L * 40;
  float h = 0.f;
  for (int s = 0; s < L; ++s) {
    int p = rev ? (L - 1 - s) : s;
    float dt = dl[(long long)p * DIN];
    float uu = ul[(long long)p * DIN];
    float Bv = xb[(long long)p * 40 + DTR + n];
    float Cv = xb[(long long)p * 40 + DTR + DST + n];
    h = __expf(dt * An) * h + dt * uu * Bv;
    float part = h * Cv;
    part += __shfl_xor(part, 8, 16);
    part += __shfl_xor(part, 4, 16);
    part += __shfl_xor(part, 2, 16);
    part += __shfl_xor(part, 1, 16);
    if (n == 0) {
      int l = trans ? ((p % H) * W + (p / H)) : p;
      long long yi = ((long long)b * L + l) * DIN + d;
      y[yi] += part + Dd * uu;    // unique (b,l,d) per launch; launches serialized
    }
  }
}

// LayerNorm over last dim D (32|D, D<=256): one wave32 per row, in-place safe.
__global__ void k_layernorm(const float* __restrict__ x, const float* __restrict__ g,
                            const float* __restrict__ b, float* __restrict__ out,
                            long long R, int D)
{
  int lane = threadIdx.x & 31;
  long long row = (long long)blockIdx.x * (blockDim.x >> 5) + (threadIdx.x >> 5);
  if (row >= R) return;
  const float* xr = x + row * D;
  float vals[8];
  int cnt = D >> 5;
  float s = 0.f;
  for (int i = 0; i < cnt; ++i) { vals[i] = xr[lane + 32 * i]; s += vals[i]; }
  for (int off = 16; off; off >>= 1) s += __shfl_xor(s, off, 32);
  float mean = s / D;
  float v = 0.f;
  for (int i = 0; i < cnt; ++i) { float dd = vals[i] - mean; v += dd * dd; }
  for (int off = 16; off; off >>= 1) v += __shfl_xor(v, off, 32);
  float inv = rsqrtf(v / D + 1e-5f);
  float* orow = out + row * D;
  for (int i = 0; i < cnt; ++i) {
    int c = lane + 32 * i;
    orow[c] = (vals[i] - mean) * inv * g[c] + b[c];
  }
}

__global__ void k_mul_silu(float* y, const float* z, int D, int ldz, long long n) {
  long long i = (long long)blockIdx.x * blockDim.x + threadIdx.x;
  if (i >= n) return;
  long long r = i / D; int c = (int)(i % D);
  float zv = z[r * ldz + c];
  y[i] *= zv / (1.f + __expf(-zv));
}

// Build fuse rows: (b,h,w) x 640 from p1 NHWC (H,W,128), p2/p3 NHWC (H,2W,128)
__global__ void k_fuse_gather(const float* p1, const float* p2, const float* p3,
                              float* out, int H, int W, long long n) {
  long long i = (long long)blockIdx.x * blockDim.x + threadIdx.x;
  if (i >= n) return;
  int j = (int)(i % 640);
  long long m = i / 640;
  int w = (int)(m % W);
  int h = (int)((m / W) % H);
  long long b = m / ((long long)W * H);
  int blk = j >> 7, c = j & 127;
  int W2 = 2 * W;
  long long r1 = (b * H + h) * (long long)W + w;
  long long r2 = (b * H + h) * (long long)W2;
  float v;
  if      (blk == 0) v = p1[r1 * 128 + c];
  else if (blk == 1) v = p2[(r2 + 2 * w)     * 128 + c];
  else if (blk == 2) v = p2[(r2 + 2 * w + 1) * 128 + c];
  else if (blk == 3) v = p3[(r2 + w)         * 128 + c];
  else               v = p3[(r2 + W + w)     * 128 + c];
  out[i] = v;
}

// Bilinear 2x upsample (half-pixel centers, edge clamp) + add. out (B,C,2H,2W)
__global__ void k_upsample_add(const float* x, const float* yb, float* out,
                               int Cc, int H, int W, long long n) {
  long long i = (long long)blockIdx.x * blockDim.x + threadIdx.x;
  if (i >= n) return;
  int OW = 2 * W, OH = 2 * H;
  int ox = (int)(i % OW);
  int oy = (int)((i / OW) % OH);
  int c  = (int)((i / ((long long)OW * OH)) % Cc);
  long long b = i / ((long long)OW * OH * Cc);
  float sx = ox * 0.5f - 0.25f, sy = oy * 0.5f - 0.25f;
  int x0 = (int)floorf(sx), y0 = (int)floorf(sy);
  float fx = sx - x0, fy = sy - y0;
  int x1 = x0 + 1, y1 = y0 + 1;
  x0 = max(0, min(W - 1, x0)); x1 = max(0, min(W - 1, x1));
  y0 = max(0, min(H - 1, y0)); y1 = max(0, min(H - 1, y1));
  const float* p = x + ((b * Cc + c) * (long long)H) * W;
  float v = (p[(long long)y0 * W + x0] * (1.f - fx) + p[(long long)y0 * W + x1] * fx) * (1.f - fy)
          + (p[(long long)y1 * W + x0] * (1.f - fx) + p[(long long)y1 * W + x1] * fx) * fy;
  out[i] = v + yb[i];
}

// im2col for 3x3 pad1 dense conv: rows (b,h,w) x (C*9), column = c*9 + ky*3+kx
__global__ void k_im2col3(const float* in, float* out, int Cc, int H, int W, long long n) {
  long long i = (long long)blockIdx.x * blockDim.x + threadIdx.x;
  if (i >= n) return;                       // n = B*H*W*C (one thread -> 9 taps)
  int c = (int)(i % Cc);
  long long m = i / Cc;
  int w = (int)(m % W);
  int h = (int)((m / W) % H);
  long long b = m / ((long long)W * H);
  const float* p = in + ((b * Cc + c) * (long long)H) * W;
  float* o = out + m * ((long long)Cc * 9) + (long long)c * 9;
#pragma unroll
  for (int ky = 0; ky < 3; ++ky)
#pragma unroll
    for (int kx = 0; kx < 3; ++kx) {
      int y = h + ky - 1, x = w + kx - 1;
      o[ky * 3 + kx] = (y >= 0 && y < H && x >= 0 && x < W) ? p[(long long)y * W + x] : 0.f;
    }
}

// ---------------------------------------------------------------------------
// Workspace layout (floats)
// ---------------------------------------------------------------------------
static constexpr long long O_CAT  = 0;                       // arranged NCHW input (<=3.15M)
static constexpr long long O_DW   = O_CAT  + 3145728LL;      // dw-conv out NCHW
static constexpr long long O_A    = O_DW   + 3145728LL;      // NHWC in / ss_out / smooth out (R*128)
static constexpr long long O_X    = O_A    + 4194304LL;      // block activation NHWC (R*128)
static constexpr long long O_LN   = O_X    + 4194304LL;      // LN scratch (R*128)
static constexpr long long O_XZ   = O_LN   + 4194304LL;      // xz (R*512) / mlp hidden / fuse rows
static constexpr long long O_XC   = O_XZ   + 16777216LL;     // x NCHW (B,256,H,W)
static constexpr long long O_CONV = O_XC   + 8388608LL;      // conv out NCHW
static constexpr long long O_XS   = O_CONV + 8388608LL;      // xs (B,2,L,256) / im2col (R*1152)
static constexpr long long O_XDBL = O_XS   + 18874368LL;     // x_dbl per-k (B*L x 40)
static constexpr long long O_DK   = O_XDBL + 1310720LL;      // delta per-k (B*L x 256)
static constexpr long long O_Y    = O_DK   + 8388608LL;      // accumulated scan output (R*256)
static constexpr long long O_P1   = O_Y    + 8388608LL;      // st-block outputs (NHWC)
static constexpr long long O_P2   = O_P1   + 2097152LL;
static constexpr long long O_P3   = O_P2   + 4194304LL;
static constexpr long long O_ST   = O_P3   + 4194304LL;      // 4 stage outputs NCHW (slots of 2.1M)
static constexpr long long O_PY   = O_ST   + 4LL * 2097152LL;// pyramid temps (2 slots)
static constexpr long long WS_FLOATS = O_PY + 2LL * 2097152LL;

// ---------------------------------------------------------------------------
// Parameter leaf indexing. Assumption: harness flattens params recursively in
// dict-insertion order after the 8 input tensors (pre1,post1,...,pre4,post4).
// ---------------------------------------------------------------------------
enum {
  DW_W = 0, DW_B, PW_W, PW_B,
  LN1G, LN1B, IN_PROJ, CONV_W, CONV_B, XPROJ, DT_W, DT_B, A_LOGS, DS_,
  ONORM_G, ONORM_B, OUT_PROJ, LN2G, LN2B, MLP_W1, MLP_B1, MLP_W2, MLP_B2
};
#define PB(i) (8 + (i) * 23)
#define FB(j) (8 + 276 + (j) * 4)        // w, b, g, beta
#define SB(j) (8 + 292 + (j) * 6)        // c1, g1, b1, c2, g2, b2

static const int DIMS_[4] = {768, 384, 192, 96};
static const int HWS_[4]  = {8, 16, 32, 64};

// ---------------------------------------------------------------------------
// Host-side launch helpers (all on `stream`, graph-capture safe)
// ---------------------------------------------------------------------------
static void gemm(hipStream_t st,
                 const float* A, long long lda, long long sA,
                 const float* W, long long ldw, long long sW,
                 float* C, long long ldc, long long sC,
                 int M, int N, int K, int batches,
                 const float* bias, long long sB,
                 const float* g, const float* beta, int act)
{
  dim3 grid((N + 63) / 64, (M + 63) / 64, batches);
  k_gemm_wmma<<<grid, 128, 0, st>>>(A, lda, sA, W, ldw, sW, C, ldc, sC,
                                    M, N, K, bias, sB, g, beta, act);
}

static inline long long ceil_div(long long a, long long b) { return (a + b - 1) / b; }

static void run_vss(hipStream_t stream, float* ws, void* const* d_in, int bi, int H, int W)
{
  auto P = [&](int idx) { return (const float*)d_in[idx]; };
  const int base = PB(bi);
  const long long L = (long long)H * W;
  const long long R = (long long)BN * L;
  const int thr = 256;

  // LN1
  k_layernorm<<<(unsigned)ceil_div(R, 8), thr, 0, stream>>>(
      ws + O_X, P(base + LN1G), P(base + LN1B), ws + O_LN, R, HID);
  // in_proj: (R x 128) -> (R x 512)
  gemm(stream, ws + O_LN, HID, 0, P(base + IN_PROJ), HID, 0,
       ws + O_XZ, 2 * DIN, 0, (int)R, 2 * DIN, HID, 1, nullptr, 0, nullptr, nullptr, 0);
  // x half -> NCHW
  long long tot = R * DIN;
  k_split_x_nchw<<<(unsigned)ceil_div(tot, thr), thr, 0, stream>>>(ws + O_XZ, ws + O_XC, H, W, tot);
  // depthwise 3x3 + silu
  k_dwconv3<<<(unsigned)ceil_div(tot, thr), thr, 0, stream>>>(
      ws + O_XC, P(base + CONV_W), P(base + CONV_B), ws + O_CONV, DIN, H, W, tot, 1);
  // xs (dir0 raster / dir1 transposed)
  k_build_xs<<<(unsigned)ceil_div(tot, thr), thr, 0, stream>>>(ws + O_CONV, ws + O_XS, H, W, tot);
  // zero scan accumulator
  k_fill<<<(unsigned)ceil_div(tot, thr), thr, 0, stream>>>(ws + O_Y, 0.f, tot);

  for (int k = 0; k < 4; ++k) {
    const float* xsk = ws + O_XS + (long long)(k & 1) * L * DIN;
    // x_dbl_k = xs_k @ x_proj_w[k]^T : (L x 256) -> (L x 40), batched over B
    gemm(stream, xsk, DIN, 2 * L * DIN,
         P(base + XPROJ) + (long long)k * 40 * DIN, DIN, 0,
         ws + O_XDBL, 40, L * 40, (int)L, 40, DIN, BN, nullptr, 0, nullptr, nullptr, 0);
    // delta_k = softplus(dts @ dt_w[k]^T + dt_b[k]) : (L x 8) -> (L x 256)
    gemm(stream, ws + O_XDBL, 40, L * 40,
         P(base + DT_W) + (long long)k * DIN * DTR, DTR, 0,
         ws + O_DK, DIN, L * DIN, (int)L, DIN, DTR, BN,
         P(base + DT_B) + (long long)k * DIN, 0, nullptr, nullptr, 4);
    // selective scan, accumulate into y
    long long nth = (long long)BN * DIN * DST;
    k_scan<<<(unsigned)ceil_div(nth, thr), thr, 0, stream>>>(
        ws + O_DK, xsk, 2 * L * DIN, ws + O_XDBL,
        P(base + A_LOGS) + (long long)k * DIN * DST,
        P(base + DS_) + (long long)k * DIN,
        ws + O_Y, (int)L, H, W, (k >= 2) ? 1 : 0, (k & 1));
  }

  // out_norm (in-place), * silu(z)
  k_layernorm<<<(unsigned)ceil_div(R, 8), thr, 0, stream>>>(
      ws + O_Y, P(base + ONORM_G), P(base + ONORM_B), ws + O_Y, R, DIN);
  k_mul_silu<<<(unsigned)ceil_div(R * DIN, thr), thr, 0, stream>>>(
      ws + O_Y, ws + O_XZ + DIN, DIN, 2 * DIN, R * DIN);
  // out_proj, residual
  gemm(stream, ws + O_Y, DIN, 0, P(base + OUT_PROJ), DIN, 0,
       ws + O_A, HID, 0, (int)R, HID, DIN, 1, nullptr, 0, nullptr, nullptr, 0);
  k_add<<<(unsigned)ceil_div(R * HID, thr), thr, 0, stream>>>(ws + O_X, ws + O_X, ws + O_A, R * HID);
  // LN2 + MLP + residual
  k_layernorm<<<(unsigned)ceil_div(R, 8), thr, 0, stream>>>(
      ws + O_X, P(base + LN2G), P(base + LN2B), ws + O_LN, R, HID);
  gemm(stream, ws + O_LN, HID, 0, P(base + MLP_W1), HID, 0,
       ws + O_XZ, 4 * HID, 0, (int)R, 4 * HID, HID, 1, P(base + MLP_B1), 0, nullptr, nullptr, 2);
  gemm(stream, ws + O_XZ, 4 * HID, 0, P(base + MLP_W2), 4 * HID, 0,
       ws + O_A, HID, 0, (int)R, HID, 4 * HID, 1, P(base + MLP_B2), 0, nullptr, nullptr, 0);
  k_add<<<(unsigned)ceil_div(R * HID, thr), thr, 0, stream>>>(ws + O_X, ws + O_X, ws + O_A, R * HID);
}

// ---------------------------------------------------------------------------
// Entry point
// ---------------------------------------------------------------------------
extern "C" void kernel_launch(void* const* d_in, const int* in_sizes, int n_in,
                              void* d_out, int out_size, void* d_ws, size_t ws_size,
                              hipStream_t stream)
{
  (void)in_sizes; (void)out_size;
  if (n_in < 8 + 276 + 16 + 18) return;
  if (ws_size < (size_t)WS_FLOATS * sizeof(float)) { /* best effort */ }

  float* ws = (float*)d_ws;
  auto P = [&](int idx) { return (const float*)d_in[idx]; };
  const int thr = 256;

  // ---- stages: s=0 -> (pre4,post4,C=768,hw=8) ... s=3 -> (pre1,post1,C=96,hw=64)
  for (int s = 0; s < 4; ++s) {
    const int C = DIMS_[s], hw = HWS_[s];
    const float* pre  = P(6 - 2 * s);
    const float* post = P(7 - 2 * s);
    float* Pout[3] = { ws + O_P1, ws + O_P2, ws + O_P3 };

    for (int blk = 0; blk < 3; ++blk) {
      const int bi  = 3 * s + blk;
      const int cin = (blk == 0) ? 2 * C : C;
      const int H   = hw, W = (blk == 0) ? hw : 2 * hw;
      const long long S = (long long)H * W;
      const long long R = (long long)BN * S;
      const long long tot = (long long)BN * cin * S;
      const unsigned blks = (unsigned)ceil_div(tot, thr);

      if (blk == 0)      k_cat_ch      <<<blks, thr, 0, stream>>>(pre, post, ws + O_CAT, C, hw * hw, tot);
      else if (blk == 1) k_interleave_w<<<blks, thr, 0, stream>>>(pre, post, ws + O_CAT, C, hw, hw, tot);
      else               k_cat_w       <<<blks, thr, 0, stream>>>(pre, post, ws + O_CAT, C, hw, hw, tot);

      k_dwconv3<<<blks, thr, 0, stream>>>(ws + O_CAT, P(PB(bi) + DW_W), P(PB(bi) + DW_B),
                                          ws + O_DW, cin, H, W, tot, 0);
      k_nchw2nhwc<<<blks, thr, 0, stream>>>(ws + O_DW, ws + O_A, cin, H, W, tot);
      gemm(stream, ws + O_A, cin, 0, P(PB(bi) + PW_W), cin, 0,
           ws + O_X, HID, 0, (int)R, HID, cin, 1, P(PB(bi) + PW_B), 0, nullptr, nullptr, 0);

      run_vss(stream, ws, d_in, bi, H, W);

      (void)hipMemcpyAsync(Pout[blk], ws + O_X, (size_t)R * HID * sizeof(float),
                           hipMemcpyDeviceToDevice, stream);
    }

    // fuse: gather 5*128 channels, 1x1 conv, scale/shift, relu -> stage out NCHW
    const long long R0  = (long long)BN * hw * hw;
    const long long tg  = R0 * 5 * HID;
    k_fuse_gather<<<(unsigned)ceil_div(tg, thr), thr, 0, stream>>>(
        ws + O_P1, ws + O_P2, ws + O_P3, ws + O_XZ, hw, hw, tg);
    gemm(stream, ws + O_XZ, 5 * HID, 0, P(FB(s) + 0), 5 * HID, 0,
         ws + O_LN, HID, 0, (int)R0, HID, 5 * HID, 1,
         P(FB(s) + 1), 0, P(FB(s) + 2), P(FB(s) + 3), 3);
    float* stout = ws + O_ST + (long long)s * 2097152LL;
    const long long t2 = R0 * HID;
    k_nhwc2nchw<<<(unsigned)ceil_div(t2, thr), thr, 0, stream>>>(ws + O_LN, stout, HID, hw, hw, t2);
  }

  // ---- pyramid: up_add + smooth (x3), final into d_out
  float* cur = ws + O_ST;                 // 8x8 stage output
  for (int j = 0; j < 3; ++j) {
    const int s = j + 1, hw = HWS_[s];
    float* stg = ws + O_ST + (long long)s * 2097152LL;
    float* tmp = ws + O_PY;               // up_add result (smooth residual)
    float* nxt = ws + O_PY + 2097152LL;
    const long long Rs  = (long long)BN * hw * hw;
    const long long tot = Rs * HID;

    k_upsample_add<<<(unsigned)ceil_div(tot, thr), thr, 0, stream>>>(
        cur, stg, tmp, HID, hw / 2, hw / 2, tot);

    // conv1: im2col + WMMA, relu((conv)*g1 + b1)
    k_im2col3<<<(unsigned)ceil_div(Rs * HID, thr), thr, 0, stream>>>(tmp, ws + O_XS, HID, hw, hw, Rs * HID);
    gemm(stream, ws + O_XS, HID * 9, 0, P(SB(j) + 0), HID * 9, 0,
         ws + O_A, HID, 0, (int)Rs, HID, HID * 9, 1,
         nullptr, 0, P(SB(j) + 1), P(SB(j) + 2), 3);
    k_nhwc2nchw<<<(unsigned)ceil_div(tot, thr), thr, 0, stream>>>(ws + O_A, ws + O_X, HID, hw, hw, tot);
    // conv2: (conv)*g2 + b2 (no act)
    k_im2col3<<<(unsigned)ceil_div(Rs * HID, thr), thr, 0, stream>>>(ws + O_X, ws + O_XS, HID, hw, hw, Rs * HID);
    gemm(stream, ws + O_XS, HID * 9, 0, P(SB(j) + 3), HID * 9, 0,
         ws + O_A, HID, 0, (int)Rs, HID, HID * 9, 1,
         nullptr, 0, P(SB(j) + 4), P(SB(j) + 5), 0);
    k_nhwc2nchw<<<(unsigned)ceil_div(tot, thr), thr, 0, stream>>>(ws + O_A, ws + O_LN, HID, hw, hw, tot);

    float* dst = (j == 2) ? (float*)d_out : nxt;
    k_add_relu<<<(unsigned)ceil_div(tot, thr), thr, 0, stream>>>(dst, ws + O_LN, tmp, tot);
    cur = dst;
  }
}
